// MultiScaleRetention_4844723110248
// MI455X (gfx1250) — compile-verified, matrix operand order checked by reading, not run on hardware
//
#include <hip/hip_runtime.h>
#include <math.h>
#include <stdint.h>

#define DEV __device__ __forceinline__

typedef __bf16 bf16;
typedef bf16        v16bf __attribute__((ext_vector_type(16)));
typedef float       v8f   __attribute__((ext_vector_type(8)));
typedef unsigned int u32x4 __attribute__((ext_vector_type(4)));
typedef int          v4i   __attribute__((ext_vector_type(4)));

// ---------------- constants ----------------
constexpr int T_LEN = 2048;
constexpr int HDm   = 2048;   // H_DIM
constexpr int NH    = 16;     // heads
constexpr int KD    = 128;    // key dim per head
constexpr int VD    = 4096;   // V_DIM
constexpr int HDIM  = 256;    // value head dim
constexpr float SCALING = 0.08838834764831845f; // 128^-0.5

// ---------------- CDNA5 async copy global -> LDS ----------------
#if __has_builtin(__builtin_amdgcn_global_load_async_to_lds_b128)
#define ASYNC_OK 1
#else
#define ASYNC_OK 0
#endif

typedef __attribute__((address_space(1))) v4i* gv4i_t;  // global v4i*
typedef __attribute__((address_space(3))) v4i* lv4i_t;  // LDS v4i*

// copy 16 bytes from global to LDS (per lane)
DEV void cp16(bf16* dst_lds, const bf16* src_glb) {
#if ASYNC_OK
  __builtin_amdgcn_global_load_async_to_lds_b128(
      (gv4i_t)(uintptr_t)src_glb,
      (lv4i_t)(unsigned)(uintptr_t)dst_lds, 0, 0);
#else
  *(u32x4*)dst_lds = *(const u32x4*)src_glb;
#endif
}

DEV void wait_async() {
#if ASYNC_OK
  asm volatile("s_wait_asynccnt 0" ::: "memory");
#endif
}

// ---------------- WMMA helpers ----------------
DEV v16bf ld16(const bf16* p) {
  union { u32x4 q[2]; v16bf v; } u;
  u.q[0] = *(const u32x4*)p;
  u.q[1] = *(const u32x4*)(p + 16);
  return u.v;
}

DEV v8f wmma_bf16(v16bf a, v16bf b, v8f c) {
  return __builtin_amdgcn_wmma_f32_16x16x32_bf16(false, a, false, b, (short)0, c,
                                                 false, false);
}

// ---------------- elementwise convert fp32 -> bf16 ----------------
__global__ void cvt_bf16_kernel(const float* __restrict__ in, bf16* __restrict__ out, int n) {
  int i = (blockIdx.x * blockDim.x + threadIdx.x) * 4;
  if (i + 3 < n) {
    out[i + 0] = (bf16)in[i + 0];
    out[i + 1] = (bf16)in[i + 1];
    out[i + 2] = (bf16)in[i + 2];
    out[i + 3] = (bf16)in[i + 3];
  }
}

// ---------------- transpose + convert: in f32 [R][C] -> out bf16 [C][R] ----------------
__global__ void transpose_cvt_kernel(const float* __restrict__ in, bf16* __restrict__ out,
                                     int R, int C) {
  __shared__ float tile[32][33];
  int c0 = blockIdx.x * 32, r0 = blockIdx.y * 32;
  for (int i = threadIdx.y; i < 32; i += 8)
    tile[i][threadIdx.x] = in[(size_t)(r0 + i) * C + c0 + threadIdx.x];
  __syncthreads();
  for (int i = threadIdx.y; i < 32; i += 8)
    out[(size_t)(c0 + i) * R + r0 + threadIdx.x] = (bf16)tile[threadIdx.x][i];
}

// ---------------- GEMM: C[M][N] = A[M][K](bf16,row) @ Bt[N][K](bf16) + bias, f32 out ---
// block 256 threads = 8 waves, 128x128 tile; wave -> 32(M) x 64(N).
// Double-buffered LDS with async global->LDS staging.
__global__ __launch_bounds__(256) void gemm_bf16_kernel(
    const bf16* __restrict__ A, const bf16* __restrict__ Bt,
    const float* __restrict__ bias, float* __restrict__ C,
    int M, int N, int K) {
  // padded row stride 40 halves (80B): 16B-aligned rows, conflict-free b128 reads
  __shared__ __align__(16) bf16 As[2][128 * 40];
  __shared__ __align__(16) bf16 Bs[2][128 * 40];

  const int tid  = threadIdx.x;
  const int lane = tid & 31;
  const int wave = tid >> 5;
  const int wm   = wave & 3;        // 0..3 -> M sub
  const int wn   = wave >> 2;       // 0..1 -> N sub
  const int l15  = lane & 15;
  const int lh   = lane >> 4;       // 0/1
  const int kb   = lh * 8;          // K-half base for operand layout
  const int m0   = blockIdx.y * 128;
  const int n0   = blockIdx.x * 128;

  v8f acc[2][4];
  v8f zero = {0.f, 0.f, 0.f, 0.f, 0.f, 0.f, 0.f, 0.f};
  for (int mi = 0; mi < 2; ++mi)
    for (int ni = 0; ni < 4; ++ni) acc[mi][ni] = zero;

  // staging lambda: tile kt -> buffer b (128 rows x 32 halves, 4 b128 per row)
  auto stage = [&](int b, int kt) {
    for (int i = 0; i < 2; ++i) {
      int idx = tid + i * 256;            // 0..511
      int r = idx >> 2, s = idx & 3;
      cp16(&As[b][r * 40 + s * 8], A + (size_t)(m0 + r) * K + kt + s * 8);
      cp16(&Bs[b][r * 40 + s * 8], Bt + (size_t)(n0 + r) * K + kt + s * 8);
#if !ASYNC_OK
      if (kt + 32 < K) {
        __builtin_prefetch(A + (size_t)(m0 + r) * K + kt + 32 + s * 8, 0, 0);
        __builtin_prefetch(Bt + (size_t)(n0 + r) * K + kt + 32 + s * 8, 0, 0);
      }
#endif
    }
  };

  stage(0, 0);
  wait_async();
  __syncthreads();

  int cur = 0;
  for (int kt = 0; kt < K; kt += 32) {
    if (kt + 32 < K) stage(cur ^ 1, kt + 32);

    v16bf a[2];
    for (int mi = 0; mi < 2; ++mi)
      a[mi] = ld16(&As[cur][(wm * 32 + mi * 16 + l15) * 40 + kb]);
    for (int ni = 0; ni < 4; ++ni) {
      v16bf b = ld16(&Bs[cur][(wn * 64 + ni * 16 + l15) * 40 + kb]);
      for (int mi = 0; mi < 2; ++mi)
        acc[mi][ni] = wmma_bf16(a[mi], b, acc[mi][ni]);
    }

    wait_async();
    __syncthreads();
    cur ^= 1;
  }

  // write out (D layout: lane holds col l15, rows lh*8 + v)
  for (int mi = 0; mi < 2; ++mi)
    for (int ni = 0; ni < 4; ++ni) {
      int col = n0 + wn * 64 + ni * 16 + l15;
      float bv = bias[col];
      for (int v = 0; v < 8; ++v) {
        int row = m0 + wm * 32 + mi * 16 + lh * 8 + v;
        C[(size_t)row * N + col] = acc[mi][ni][v] + bv;
      }
    }
}

// ---------------- rotary (theta_shift), writes per-head bf16 [head][T][128] ----------
__global__ void rotary_kernel(const float* __restrict__ x,
                              const float* __restrict__ sinp,
                              const float* __restrict__ cosp,
                              bf16* __restrict__ out, float scale) {
  int gid  = blockIdx.x * blockDim.x + threadIdx.x; // T*16*64 pairs
  int i    = gid & 63;
  int head = (gid >> 6) & 15;
  int t    = gid >> 10;
  int d    = i * 2;
  size_t xi = (size_t)t * HDm + head * KD + d;
  float x0 = x[xi] * scale;
  float x1 = x[xi + 1] * scale;
  float c0 = cosp[t * KD + d], c1 = cosp[t * KD + d + 1];
  float s0 = sinp[t * KD + d], s1 = sinp[t * KD + d + 1];
  size_t o = ((size_t)head * T_LEN + t) * KD + d;
  out[o]     = (bf16)(x0 * c0 - x1 * s0);
  out[o + 1] = (bf16)(x1 * c1 + x0 * s1);
}

// ---------------- retention: flash-style single pass ----------------
// grid (32 row-blocks, 16 heads), block 128 = 4 waves; wave -> 16 rows.
// O[t][head*256+ch] = (1/max(|rowsum|,1)) * sum_t' qr.kr^T * mask * V
__global__ __launch_bounds__(128) void retention_kernel(
    const bf16* __restrict__ qr,   // [16][2048][128]
    const bf16* __restrict__ kr,   // [16][2048][128]
    const bf16* __restrict__ vt,   // [4096][2048]  (ch-major, t contiguous)
    const float* __restrict__ mask,// [16][2048][2048]
    float* __restrict__ O) {       // [2048][4096]
  __shared__ __align__(16) bf16 Ks[32 * 136];   // kr tile, stride 272B
  __shared__ __align__(16) bf16 Vs[256 * 40];   // v^T tile, stride 80B
  __shared__ __align__(16) bf16 Ps[4 * 16 * 40];// per-wave P patch

  const int tid   = threadIdx.x;
  const int lane  = tid & 31;
  const int wave  = tid >> 5;
  const int l15   = lane & 15;
  const int lh    = lane >> 4;
  const int kb    = lh * 8;
  const int rhalf = lh * 8;
  const int head  = blockIdx.y;
  const int rb    = blockIdx.x;
  const int rbase = rb * 64 + wave * 16;

  // load this wave's 16 qr rows as 4 A-operands (K=128 -> 4 k-steps of 32)
  v16bf qa[4];
  for (int kk = 0; kk < 4; ++kk)
    qa[kk] = ld16(qr + ((size_t)(head * T_LEN + rbase + l15)) * KD + kk * 32 + kb);

  v8f zero = {0.f, 0.f, 0.f, 0.f, 0.f, 0.f, 0.f, 0.f};
  v8f accO[16];
  for (int ni = 0; ni < 16; ++ni) accO[ni] = zero;
  float rs[8] = {0.f, 0.f, 0.f, 0.f, 0.f, 0.f, 0.f, 0.f};

  const int ctn = rb * 2 + 2;  // causal: only column tiles up to block diagonal
  for (int ct = 0; ct < ctn; ++ct) {
    __syncthreads();           // previous tile fully consumed by all waves
    const int c0 = ct * 32;
    // stage kr tile: 32 rows x 128 halves (async DMA to LDS)
    for (int i = 0; i < 4; ++i) {
      int idx = tid + i * 128;  // 0..511
      int r = idx >> 4, s = idx & 15;
      cp16(&Ks[r * 136 + s * 8],
           kr + ((size_t)(head * T_LEN + c0 + r)) * KD + s * 8);
    }
    // stage v^T tile: 256 ch-rows x 32 halves
    for (int i = 0; i < 8; ++i) {
      int idx = tid + i * 128;  // 0..1023
      int r = idx >> 2, s = idx & 3;
      cp16(&Vs[r * 40 + s * 8],
           vt + ((size_t)(head * HDIM + r)) * T_LEN + c0 + s * 8);
    }
    wait_async();
    __syncthreads();

    bf16* wp = &Ps[wave * 16 * 40];
    for (int nsub = 0; nsub < 2; ++nsub) {
      v8f s = zero;
      for (int kk = 0; kk < 4; ++kk) {
        v16bf b = ld16(&Ks[(nsub * 16 + l15) * 136 + kk * 32 + kb]);
        s = wmma_bf16(qa[kk], b, s);
      }
      // mask multiply + rowsum + store P (bf16) for second WMMA
      int col = c0 + nsub * 16 + l15;
      const float* mrow =
          mask + ((size_t)head * T_LEN + rbase + rhalf) * T_LEN + col;
      for (int v = 0; v < 8; ++v) {
        float pv = s[v] * mrow[(size_t)v * T_LEN];
        rs[v] += pv;
        wp[(rhalf + v) * 40 + nsub * 16 + l15] = (bf16)pv;
      }
    }
    // O += P @ V  (K = 32 columns of this tile)
    v16bf pa = ld16(&wp[l15 * 40 + kb]);
    for (int ni = 0; ni < 16; ++ni) {
      v16bf b = ld16(&Vs[(ni * 16 + l15) * 40 + kb]);
      accO[ni] = wmma_bf16(pa, b, accO[ni]);
    }
  }

  // reduce rowsums across the 16 lanes sharing each row (keep lane-halves apart)
  for (int v = 0; v < 8; ++v) {
    float x = rs[v];
    x += __shfl_xor(x, 1, 16);
    x += __shfl_xor(x, 2, 16);
    x += __shfl_xor(x, 4, 16);
    x += __shfl_xor(x, 8, 16);
    rs[v] = 1.f / fmaxf(fabsf(x), 1.f);
  }
  for (int ni = 0; ni < 16; ++ni)
    for (int v = 0; v < 8; ++v) {
      int row = rbase + rhalf + v;
      O[(size_t)row * VD + head * HDIM + ni * 16 + l15] = accO[ni][v] * rs[v];
    }
}

// ---------------- groupnorm(256) + silu gate -> bf16 ----------------
__global__ __launch_bounds__(256) void gn_gate_kernel(
    const float* __restrict__ O, const float* __restrict__ g,
    bf16* __restrict__ out) {
  __shared__ float s1[256];
  __shared__ float s2[256];
  int th = blockIdx.x;         // t*16 + head
  int t = th >> 4, head = th & 15;
  int ch = threadIdx.x;
  size_t idx = (size_t)t * VD + head * HDIM + ch;
  float x = O[idx];
  s1[ch] = x;
  s2[ch] = x * x;
  __syncthreads();
  for (int off = 128; off > 0; off >>= 1) {
    if (ch < off) { s1[ch] += s1[ch + off]; s2[ch] += s2[ch + off]; }
    __syncthreads();
  }
  float mu  = s1[0] * (1.f / 256.f);
  float var = s2[0] * (1.f / 256.f) - mu * mu;
  float nrm = (x - mu) * rsqrtf(var + 1e-6f);
  float gv  = g[idx];
  float sil = gv / (1.f + __expf(-gv));
  out[idx] = (bf16)(sil * nrm);
}

// ---------------- workspace layout (bytes) ----------------
constexpr size_t MB = 1ull << 20;
constexpr size_t OFS_HSB = 0;          // hs bf16                8 MB
constexpr size_t OFS_WQT = 8  * MB;    // Wq^T bf16              8 MB
constexpr size_t OFS_WKT = 16 * MB;    // Wk^T bf16              8 MB
constexpr size_t OFS_WVT = 24 * MB;    // Wv^T bf16             16 MB
constexpr size_t OFS_WGT = 40 * MB;    // Wg^T bf16             16 MB
constexpr size_t OFS_WOT = 56 * MB;    // Wo^T bf16             16 MB
constexpr size_t OFS_QF  = 72 * MB;    // q f32                 16 MB
constexpr size_t OFS_KF  = 88 * MB;    // k f32                 16 MB
constexpr size_t OFS_VF  = 104 * MB;   // v f32                 32 MB
constexpr size_t OFS_GF  = 136 * MB;   // g f32                 32 MB
constexpr size_t OFS_QR  = 168 * MB;   // qr bf16                8 MB
constexpr size_t OFS_KR  = 176 * MB;   // kr bf16                8 MB
constexpr size_t OFS_VT  = 184 * MB;   // v^T bf16              16 MB
constexpr size_t OFS_O   = 72 * MB;    // O f32 (reuses q/k)    32 MB
constexpr size_t OFS_GTD = 104 * MB;   // gated bf16 (reuses v) 16 MB

extern "C" void kernel_launch(void* const* d_in, const int* in_sizes, int n_in,
                              void* d_out, int out_size, void* d_ws, size_t ws_size,
                              hipStream_t stream) {
  (void)in_sizes; (void)n_in; (void)out_size; (void)ws_size;
  const float* hs   = (const float*)d_in[0];
  const float* Wq   = (const float*)d_in[1];
  const float* bq   = (const float*)d_in[2];
  const float* Wk   = (const float*)d_in[3];
  const float* bk   = (const float*)d_in[4];
  const float* Wv   = (const float*)d_in[5];
  const float* bv   = (const float*)d_in[6];
  const float* Wg   = (const float*)d_in[7];
  const float* bg   = (const float*)d_in[8];
  const float* Wo   = (const float*)d_in[9];
  const float* bo   = (const float*)d_in[10];
  const float* sinp = (const float*)d_in[11];
  const float* cosp = (const float*)d_in[12];
  const float* dmsk = (const float*)d_in[13];
  float* out = (float*)d_out;

  char* ws = (char*)d_ws;
  bf16*  HSB = (bf16*)(ws + OFS_HSB);
  bf16*  WQT = (bf16*)(ws + OFS_WQT);
  bf16*  WKT = (bf16*)(ws + OFS_WKT);
  bf16*  WVT = (bf16*)(ws + OFS_WVT);
  bf16*  WGT = (bf16*)(ws + OFS_WGT);
  bf16*  WOT = (bf16*)(ws + OFS_WOT);
  float* QF  = (float*)(ws + OFS_QF);
  float* KF  = (float*)(ws + OFS_KF);
  float* VF  = (float*)(ws + OFS_VF);
  float* GF  = (float*)(ws + OFS_GF);
  bf16*  QR  = (bf16*)(ws + OFS_QR);
  bf16*  KR  = (bf16*)(ws + OFS_KR);
  bf16*  VT  = (bf16*)(ws + OFS_VT);
  float* OB  = (float*)(ws + OFS_O);
  bf16*  GTD = (bf16*)(ws + OFS_GTD);

  // 1) convert hidden_states to bf16
  cvt_bf16_kernel<<<(T_LEN * HDm / 4 + 255) / 256, 256, 0, stream>>>(hs, HSB, T_LEN * HDm);

  // 2) weight transposes to bf16 [N][K]
  dim3 tb(32, 8);
  transpose_cvt_kernel<<<dim3(HDm / 32, HDm / 32), tb, 0, stream>>>(Wq, WQT, HDm, HDm);
  transpose_cvt_kernel<<<dim3(HDm / 32, HDm / 32), tb, 0, stream>>>(Wk, WKT, HDm, HDm);
  transpose_cvt_kernel<<<dim3(VD  / 32, HDm / 32), tb, 0, stream>>>(Wv, WVT, HDm, VD);
  transpose_cvt_kernel<<<dim3(VD  / 32, HDm / 32), tb, 0, stream>>>(Wg, WGT, HDm, VD);
  transpose_cvt_kernel<<<dim3(HDm / 32, VD  / 32), tb, 0, stream>>>(Wo, WOT, VD, HDm);

  // 3) projection GEMMs (bf16 WMMA, f32 out)
  gemm_bf16_kernel<<<dim3(HDm / 128, T_LEN / 128), 256, 0, stream>>>(HSB, WQT, bq, QF, T_LEN, HDm, HDm);
  gemm_bf16_kernel<<<dim3(HDm / 128, T_LEN / 128), 256, 0, stream>>>(HSB, WKT, bk, KF, T_LEN, HDm, HDm);
  gemm_bf16_kernel<<<dim3(VD  / 128, T_LEN / 128), 256, 0, stream>>>(HSB, WVT, bv, VF, T_LEN, VD, HDm);
  gemm_bf16_kernel<<<dim3(VD  / 128, T_LEN / 128), 256, 0, stream>>>(HSB, WGT, bg, GF, T_LEN, VD, HDm);

  // 4) rotary embedding -> per-head bf16 qr/kr (k pre-scaled by 128^-0.5)
  int rot_blocks = T_LEN * NH * (KD / 2) / 256;
  rotary_kernel<<<rot_blocks, 256, 0, stream>>>(QF, sinp, cosp, QR, 1.0f);
  rotary_kernel<<<rot_blocks, 256, 0, stream>>>(KF, sinp, cosp, KR, SCALING);

  // 5) transpose V: f32 [T][4096] -> bf16 [4096][T]
  transpose_cvt_kernel<<<dim3(VD / 32, T_LEN / 32), tb, 0, stream>>>(VF, VT, T_LEN, VD);

  // 6) retention (flash-style, mask fused, post-hoc row normalization)
  retention_kernel<<<dim3(T_LEN / 64, NH), 128, 0, stream>>>(QR, KR, VT, dmsk, OB);

  // 7) groupnorm + silu gate
  gn_gate_kernel<<<T_LEN * NH, 256, 0, stream>>>(OB, GF, GTD);

  // 8) output projection GEMM -> d_out
  gemm_bf16_kernel<<<dim3(HDm / 128, T_LEN / 128), 256, 0, stream>>>(GTD, WOT, bo, out, T_LEN, HDm, VD);
}
